// AurelianMemoryCore_34677565948787
// MI455X (gfx1250) — compile-verified, matrix operand order checked by reading, not run on hardware
//
#include <hip/hip_runtime.h>
#include <hip/hip_bf16.h>

// ---------------------------------------------------------------------------
// Types for CDNA5 WMMA
// ---------------------------------------------------------------------------
typedef __bf16 bf16;
typedef __attribute__((ext_vector_type(16))) __bf16 bf16x16;
typedef __attribute__((ext_vector_type(8)))  float  f32x8;

#define ACT_NONE 0
#define ACT_SIGMOID 1

static __device__ __forceinline__ f32x8 wmma_bf16(bf16x16 a, bf16x16 b, f32x8 c) {
    // v_wmma_f32_16x16x32_bf16: D = A(16x32) * B(32x16) + C(16x16 f32)
    return __builtin_amdgcn_wmma_f32_16x16x32_bf16(
        /*neg_a=*/false, a, /*neg_b=*/false, b,
        /*c_mod=*/(short)0, c, /*reuse_a=*/false, /*reuse_b=*/false);
}

// Load a 16x32 bf16 fragment from global, row-major source with leading dim ld.
// A-operand semantics: row M = lane&15; K packing: lanes 0-15 hold K 0..7 and
// 16..23, lanes 16-31 hold K 8..15 and 24..31 (ISA 7.12.2). The same loader
// serves B operands when W is stored row-major [cols][K] (column-per-lane = W
// row per lane, identical K packing).
static __device__ __forceinline__ bf16x16 load_frag(const bf16* __restrict__ p0,
                                                    long ld, int row0, int k0) {
    const int l = threadIdx.x & 31;
    const bf16* p = p0 + (long)(row0 + (l & 15)) * ld + (k0 + ((l >> 4) << 3));
    bf16x16 f;
    uint4* fp = reinterpret_cast<uint4*>(&f);
    fp[0] = *reinterpret_cast<const uint4*>(p);        // K: base .. base+7
    fp[1] = *reinterpret_cast<const uint4*>(p + 16);   // K: base+16 .. base+23
    return f;
}

// Fragment load from a per-lane base pointer + compile-time byte offset so the
// offset folds into the VMEM instruction's 24-bit IOFFSET.
static __device__ __forceinline__ bf16x16 load_frag_off(const bf16* p,
                                                        int byteoff) {
    bf16x16 f;
    uint4* fp = reinterpret_cast<uint4*>(&f);
    fp[0] = *reinterpret_cast<const uint4*>((const char*)p + byteoff);
    fp[1] = *reinterpret_cast<const uint4*>((const char*)p + byteoff + 32);
    return f;
}

// Same fragment gather but from an LDS bf16 buffer with row stride 128.
static __device__ __forceinline__ bf16x16 load_frag_lds(const bf16* sP,
                                                        int row, int koff) {
    const bf16* p = sP + row * 128 + koff;
    bf16x16 f;
    uint4* fp = reinterpret_cast<uint4*>(&f);
    fp[0] = *reinterpret_cast<const uint4*>(p);        // ds_load_b128
    fp[1] = *reinterpret_cast<const uint4*>(p + 16);   // ds_load_b128
    return f;
}

// ---------------------------------------------------------------------------
// Generic GEMM: out[n_rows x n_cols](f32) = A[n_rows x K](bf16) * W[n_cols x K]^T
//   + bias, optional accumulate (beta), optional sigmoid.
// One wave computes a 16x64 tile (4 WMMA accumulators). The K loop processes
// 64 K per iteration with two ping-ponged fragment buffers: each buffer is
// refilled in place right after its WMMA burst, so there are no register
// copies, and each 10-load clause is covered by the other buffer's 4 WMMAs of latency.
// All K values used here are multiples of 64.
// ---------------------------------------------------------------------------
__global__ __launch_bounds__(256) void gemm_bf16_kernel(
    const bf16* __restrict__ A, long lda,
    const bf16* __restrict__ W, long ldw,
    const float* __restrict__ bias,
    float* __restrict__ out, long ldo,
    int n_rows, int n_cols, int K, int beta, int act)
{
    const int wave      = blockIdx.x * 8 + (threadIdx.x >> 5);
    const int col_tiles = n_cols >> 6;
    const int row_tile  = wave / col_tiles;
    const int col_tile  = wave % col_tiles;
    if (row_tile >= (n_rows >> 4)) return;
    const int row0 = row_tile << 4;
    const int col0 = col_tile << 6;
    const int l    = threadIdx.x & 31;
    const int l15  = l & 15;
    const int lk   = (l >> 4) << 3;

    // Per-lane base pointers (advanced by 64 elements / 128 bytes per iter).
    const bf16* pa  = A + (long)(row0 + l15) * lda + lk;
    const bf16* pw0 = W + (long)(col0 +  0 + l15) * ldw + lk;
    const bf16* pw1 = W + (long)(col0 + 16 + l15) * ldw + lk;
    const bf16* pw2 = W + (long)(col0 + 32 + l15) * ldw + lk;
    const bf16* pw3 = W + (long)(col0 + 48 + l15) * ldw + lk;

    const f32x8 zero = {0.f, 0.f, 0.f, 0.f, 0.f, 0.f, 0.f, 0.f};
    f32x8 acc[4];
#pragma unroll
    for (int j = 0; j < 4; ++j) acc[j] = zero;

    // Prologue: buffers for k = 0 (byte off 0) and k = 32 (byte off 64).
    bf16x16 a0 = load_frag_off(pa, 0);
    bf16x16 b0[4] = { load_frag_off(pw0, 0), load_frag_off(pw1, 0),
                      load_frag_off(pw2, 0), load_frag_off(pw3, 0) };
    bf16x16 a1 = load_frag_off(pa, 64);
    bf16x16 b1[4] = { load_frag_off(pw0, 64), load_frag_off(pw1, 64),
                      load_frag_off(pw2, 64), load_frag_off(pw3, 64) };

    for (int k = 0; k + 64 < K; k += 64) {
        // consume buf0 (k), refill buf0 with k+64 (byte off 128)
#pragma unroll
        for (int j = 0; j < 4; ++j) acc[j] = wmma_bf16(a0, b0[j], acc[j]);
        a0    = load_frag_off(pa, 128);
        b0[0] = load_frag_off(pw0, 128);
        b0[1] = load_frag_off(pw1, 128);
        b0[2] = load_frag_off(pw2, 128);
        b0[3] = load_frag_off(pw3, 128);
        // consume buf1 (k+32), refill buf1 with k+96 (byte off 192)
#pragma unroll
        for (int j = 0; j < 4; ++j) acc[j] = wmma_bf16(a1, b1[j], acc[j]);
        a1    = load_frag_off(pa, 192);
        b1[0] = load_frag_off(pw0, 192);
        b1[1] = load_frag_off(pw1, 192);
        b1[2] = load_frag_off(pw2, 192);
        b1[3] = load_frag_off(pw3, 192);
        pa += 64; pw0 += 64; pw1 += 64; pw2 += 64; pw3 += 64;
    }
    // Tail: last two k-steps already resident in the buffers.
#pragma unroll
    for (int j = 0; j < 4; ++j) acc[j] = wmma_bf16(a0, b0[j], acc[j]);
#pragma unroll
    for (int j = 0; j < 4; ++j) acc[j] = wmma_bf16(a1, b1[j], acc[j]);

    // C/D layout: lane l, VGPR v -> row = (l>>4)*8 + v, col = l&15
    const int mbase = (l >> 4) << 3;
#pragma unroll
    for (int j = 0; j < 4; ++j) {
        const int col = col0 + j * 16 + l15;
#pragma unroll
        for (int v = 0; v < 8; ++v) {
            long idx = (long)(row0 + mbase + v) * ldo + col;
            float x  = acc[j][v];
            if (bias) x += bias[col];
            if (beta) x += out[idx];
            if (act == ACT_SIGMOID) x = 1.0f / (1.0f + __expf(-x));
            out[idx] = x;
        }
    }
}

// ---------------------------------------------------------------------------
// Flash-attention over the memory bank.
//   q     : [N, 512]  bf16, pre-scaled by 1/sqrt(512)
//   memb  : [8192, 512] bf16 (K operand, row-major)
//   memT  : [512, 8192] bf16 (V^T, so B fragments of V load contiguously)
//   mread : [N, 512]  f32
// One workgroup (8 waves / 256 threads) handles 16 query rows. Capacity is
// processed in 128-slot tiles: wave w computes S tile columns w*16..w*16+15,
// online softmax is reduced through LDS (P stored to LDS as bf16 in fragment-
// friendly layout), then wave w owns output columns w*64..w*64+63 for P@V.
// ---------------------------------------------------------------------------
#define CAP 8192
#define DM  512

__global__ __launch_bounds__(256) void attn_kernel(
    const bf16* __restrict__ q,
    const bf16* __restrict__ memb,
    const bf16* __restrict__ memT,
    float* __restrict__ mread)
{
    __shared__ float sS[16][132];                   // raw logits (padded)
    __shared__ __align__(16) bf16 sP[16][128];      // probabilities, bf16
    __shared__ float pmax[16][16];
    __shared__ float psum[16][16];
    __shared__ float rowmax[16], rowsum[16], ralpha[16];

    const int tid  = threadIdx.x;
    const int w    = tid >> 5;
    const int l    = tid & 31;
    const int row0 = blockIdx.x << 4;

    // Preload the 16x512 Q block as 16 fragments (kept in VGPRs).
    bf16x16 qf[16];
#pragma unroll
    for (int kk = 0; kk < 16; ++kk) qf[kk] = load_frag(q, DM, row0, kk * 32);

    const f32x8 zero = {0.f, 0.f, 0.f, 0.f, 0.f, 0.f, 0.f, 0.f};
    f32x8 o[4];
#pragma unroll
    for (int j = 0; j < 4; ++j) o[j] = zero;

    if (tid < 16) { rowmax[tid] = -3.0e38f; rowsum[tid] = 0.0f; }
    __syncthreads();

    const int r     = tid >> 4;        // stats: row handled by this thread
    const int ch    = tid & 15;        // stats: 8-wide chunk within row
    const int sbase = ch * 8;
    const int n15   = l & 15;
    const int mb    = (l >> 4) << 3;

    for (int ct = 0; ct < CAP; ct += 128) {
        // ---- S = Q * K^T  (this wave: capacity rows c0..c0+15) ----
        // Ping-pong buffered over the 16 k-steps (fully unrolled: renames).
        const int c0 = ct + w * 16;
        f32x8 s = zero;
        bf16x16 kc0 = load_frag(memb, DM, c0, 0);
        bf16x16 kc1 = load_frag(memb, DM, c0, 32);
#pragma unroll
        for (int kk = 0; kk < 16; kk += 2) {
            s = wmma_bf16(qf[kk], kc0, s);
            if (kk + 2 < 16) kc0 = load_frag(memb, DM, c0, (kk + 2) * 32);
            s = wmma_bf16(qf[kk + 1], kc1, s);
            if (kk + 3 < 16) kc1 = load_frag(memb, DM, c0, (kk + 3) * 32);
        }
#pragma unroll
        for (int v = 0; v < 8; ++v) sS[mb + v][w * 16 + n15] = s[v];
        __syncthreads();

        // ---- online softmax stats across the 16x128 tile ----
        float pm = -3.0e38f;
#pragma unroll
        for (int kx = 0; kx < 8; ++kx) pm = fmaxf(pm, sS[r][sbase + kx]);
        pmax[r][ch] = pm;
        __syncthreads();
        if (tid < 16) {
            float mnew = rowmax[tid];
            for (int c = 0; c < 16; ++c) mnew = fmaxf(mnew, pmax[tid][c]);
            ralpha[tid] = __expf(rowmax[tid] - mnew);
            rowmax[tid] = mnew;
        }
        __syncthreads();
        const float mrow = rowmax[r];
        float ps = 0.0f;
#pragma unroll
        for (int kx = 0; kx < 8; ++kx) {
            float p = __expf(sS[r][sbase + kx] - mrow);
            sP[r][sbase + kx] = (bf16)p;     // fragment-ready bf16 P
            ps += p;
        }
        psum[r][ch] = ps;
        __syncthreads();
        if (tid < 16) {
            float sacc = rowsum[tid] * ralpha[tid];
            for (int c = 0; c < 16; ++c) sacc += psum[tid][c];
            rowsum[tid] = sacc;
        }
        __syncthreads();

        // ---- rescale O accumulators by alpha(row) ----
        float av[8];
#pragma unroll
        for (int v = 0; v < 8; ++v) av[v] = ralpha[mb + v];
#pragma unroll
        for (int j = 0; j < 4; ++j)
#pragma unroll
            for (int v = 0; v < 8; ++v) o[j][v] *= av[v];

        // ---- O += P * V   (this wave: V columns w*64 .. w*64+63) ----
#pragma unroll
        for (int kt = 0; kt < 4; ++kt) {
            bf16x16 vf[4];
#pragma unroll
            for (int j = 0; j < 4; ++j)
                vf[j] = load_frag(memT, CAP, w * 64 + j * 16, ct + kt * 32);
            bf16x16 pf = load_frag_lds(&sP[0][0], n15, kt * 32 + mb);
#pragma unroll
            for (int j = 0; j < 4; ++j) o[j] = wmma_bf16(pf, vf[j], o[j]);
        }
        __syncthreads();   // sS / sP reused next iteration
    }

    // ---- epilogue: normalize and store ----
    float inv[8];
#pragma unroll
    for (int v = 0; v < 8; ++v) inv[v] = 1.0f / rowsum[mb + v];
#pragma unroll
    for (int j = 0; j < 4; ++j) {
        const int col = w * 64 + j * 16 + n15;
#pragma unroll
        for (int v = 0; v < 8; ++v)
            mread[(long)(row0 + mb + v) * DM + col] = o[j][v] * inv[v];
    }
}

// ---------------------------------------------------------------------------
// Elementwise helpers
// ---------------------------------------------------------------------------
struct bf16x4 { bf16 v[4]; };   // 8-byte packed store

__global__ __launch_bounds__(256) void cvt4_kernel(const float4* __restrict__ src,
                                                   bf16x4* __restrict__ dst,
                                                   float scale, int n4) {
    int i = blockIdx.x * 256 + threadIdx.x;
    if (i < n4) {
        float4 x = src[i];
        bf16x4 y;
        y.v[0] = (bf16)(x.x * scale);
        y.v[1] = (bf16)(x.y * scale);
        y.v[2] = (bf16)(x.z * scale);
        y.v[3] = (bf16)(x.w * scale);
        dst[i] = y;
    }
}

__global__ __launch_bounds__(256) void transpose_cvt_kernel(
    const float* __restrict__ mem, bf16* __restrict__ memT) {
    int i = blockIdx.x * 256 + threadIdx.x;   // over 512*8192
    int c = i & (CAP - 1);
    int m = i >> 13;
    if (m < DM) memT[(long)m * CAP + c] = (bf16)mem[(long)c * DM + m];
}

__global__ __launch_bounds__(256) void gate_mul_kernel(
    const float* __restrict__ mread, const float* __restrict__ fgt,
    float* __restrict__ gatedf, bf16* __restrict__ gatedb, int n) {
    int i = blockIdx.x * 256 + threadIdx.x;
    if (i < n) {
        float g   = mread[i] * fgt[i];
        gatedf[i] = g;
        gatedb[i] = (bf16)g;
    }
}

__global__ __launch_bounds__(256) void mulcvt_kernel(
    const float* __restrict__ a, const float* __restrict__ b,
    bf16* __restrict__ dst, int n) {
    int i = blockIdx.x * 256 + threadIdx.x;
    if (i < n) dst[i] = (bf16)(a[i] * b[i]);
}

__global__ __launch_bounds__(256) void add_res_kernel(
    float* __restrict__ out, const float* __restrict__ h, int n) {
    int i = blockIdx.x * 256 + threadIdx.x;
    if (i < n) out[i] += h[i];
}

// ---------------------------------------------------------------------------
// Host-side pipeline
// ---------------------------------------------------------------------------
extern "C" void kernel_launch(void* const* d_in, const int* in_sizes, int n_in,
                              void* d_out, int out_size, void* d_ws, size_t ws_size,
                              hipStream_t stream) {
    (void)in_sizes; (void)n_in; (void)out_size; (void)ws_size;
    const int N = 8192;      // B*T
    const int D = 2048;      // d_model
    const int M = 512;       // d_mem
    const int C = 8192;      // capacity
    const int KG = D + M;    // 2560

    const float* h        = (const float*)d_in[0];
    const float* q_w      = (const float*)d_in[1];
    const float* q_b      = (const float*)d_in[2];
    const float* mem      = (const float*)d_in[3];
    const float* forget_w = (const float*)d_in[4];
    const float* forget_b = (const float*)d_in[5];
    const float* go_w     = (const float*)d_in[6];
    const float* go_b     = (const float*)d_in[7];
    const float* out_w    = (const float*)d_in[8];
    const float* out_b    = (const float*)d_in[9];
    float* out = (float*)d_out;

    char* wsb = (char*)d_ws;
    size_t off = 0;
    auto alloc = [&](size_t bytes) {
        size_t o = off;
        off = (off + bytes + 255) & ~(size_t)255;
        return o;
    };
    bf16*  hb     = (bf16*)(wsb + alloc((size_t)N * D * 2));
    bf16*  memb   = (bf16*)(wsb + alloc((size_t)C * M * 2));
    bf16*  memT   = (bf16*)(wsb + alloc((size_t)M * C * 2));
    bf16*  qwb    = (bf16*)(wsb + alloc((size_t)M * D * 2));
    bf16*  fwb    = (bf16*)(wsb + alloc((size_t)M * D * 2));
    bf16*  gowb   = (bf16*)(wsb + alloc((size_t)M * KG * 2));
    bf16*  owb    = (bf16*)(wsb + alloc((size_t)D * M * 2));
    float* qf32   = (float*)(wsb + alloc((size_t)N * M * 4));  // reused as gate logits
    bf16*  qb     = (bf16*)(wsb + alloc((size_t)N * M * 2));
    float* fgt    = (float*)(wsb + alloc((size_t)N * M * 4));
    float* mreadp = (float*)(wsb + alloc((size_t)N * M * 4));
    float* gatedf = (float*)(wsb + alloc((size_t)N * M * 4));
    bf16*  gatedb = (bf16*)(wsb + alloc((size_t)N * M * 2));
    bf16*  ub     = (bf16*)(wsb + alloc((size_t)N * M * 2));
    float* gl     = qf32;   // reuse: q f32 is dead after the bf16 convert

    auto grid1d = [](int n) { return dim3((unsigned)((n + 255) / 256)); };
    auto cvt4 = [&](const float* s, bf16* d, float sc, int n) {
        int n4 = n / 4;
        cvt4_kernel<<<grid1d(n4), 256, 0, stream>>>(
            (const float4*)s, (bf16x4*)d, sc, n4);
    };
    auto gemm_blocks = [](int rows, int cols) {
        return dim3((unsigned)(((rows / 16) * (cols / 64) + 7) / 8));
    };

    // 1) bf16 conversions
    cvt4(h, hb, 1.0f, N * D);
    cvt4(q_w, qwb, 1.0f, M * D);
    cvt4(forget_w, fwb, 1.0f, M * D);
    cvt4(go_w, gowb, 1.0f, M * KG);
    cvt4(out_w, owb, 1.0f, D * M);
    cvt4(mem, memb, 1.0f, C * M);
    transpose_cvt_kernel<<<grid1d(M * C), 256, 0, stream>>>(mem, memT);

    // 2) q = h @ q_w^T + q_b ; then fold softmax scale into bf16 q
    gemm_bf16_kernel<<<gemm_blocks(N, M), 256, 0, stream>>>(
        hb, D, qwb, D, q_b, qf32, M, N, M, D, 0, ACT_NONE);
    const float scale = 0.044194173824159216f;  // 1/sqrt(512)
    cvt4(qf32, qb, scale, N * M);

    // 3) forget = sigmoid(h @ forget_w^T + forget_b)
    gemm_bf16_kernel<<<gemm_blocks(N, M), 256, 0, stream>>>(
        hb, D, fwb, D, forget_b, fgt, M, N, M, D, 0, ACT_SIGMOID);

    // 4) mem_read = softmax(q mem^T) mem
    attn_kernel<<<dim3(N / 16), 256, 0, stream>>>(qb, memb, memT, mreadp);

    // 5) gated_mem = mem_read * forget (f32 + bf16 copies)
    gate_mul_kernel<<<grid1d(N * M), 256, 0, stream>>>(mreadp, fgt, gatedf,
                                                       gatedb, N * M);

    // 6) gate_w = sigmoid(h @ go_w[:, :D]^T + gated @ go_w[:, D:]^T + go_b)
    gemm_bf16_kernel<<<gemm_blocks(N, M), 256, 0, stream>>>(
        hb, D, gowb, KG, go_b, gl, M, N, M, D, 0, ACT_NONE);
    gemm_bf16_kernel<<<gemm_blocks(N, M), 256, 0, stream>>>(
        gatedb, M, gowb + D, KG, nullptr, gl, M, N, M, M, 1, ACT_SIGMOID);

    // 7) u = gate_w * gated_mem (bf16)
    mulcvt_kernel<<<grid1d(N * M), 256, 0, stream>>>(gl, gatedf, ub, N * M);

    // 8) out = u @ out_w^T + out_b ; out += h
    gemm_bf16_kernel<<<gemm_blocks(N, D), 256, 0, stream>>>(
        ub, M, owb, M, out_b, out, D, N, D, M, 0, ACT_NONE);
    add_res_kernel<<<grid1d(N * D), 256, 0, stream>>>(out, h, N * D);
}
